// GPT_69389491634268
// MI455X (gfx1250) — matrix operand order repaired and checked by reading, not compiled, not run on hardware
//
#include <hip/hip_runtime.h>

// Model constants (match reference)
#define C_DIM 1024
#define T_DIM 1024
#define B_DIM 4
#define H_DIM 16
#define D_DIM 64
#define L_DIM 6
#define V_DIM 512
#define FF_DIM 4096
#define BT (B_DIM * T_DIM)

typedef __attribute__((ext_vector_type(16))) __bf16 v16bf;
typedef __attribute__((ext_vector_type(8)))  __bf16 v8bf;
typedef __attribute__((ext_vector_type(8)))  float  v8f;

#define WMMA_BF16(A, B, Cc) \
  __builtin_amdgcn_wmma_f32_16x16x32_bf16(false, (A), false, (B), (short)0, (Cc), false, false)

__device__ __forceinline__ unsigned short f32_to_bf16(float f) {
  union { float f; unsigned u; } cv; cv.f = f;
  unsigned u = cv.u;
  return (unsigned short)((u + 0x7fffu + ((u >> 16) & 1u)) >> 16);
}

// A-style fragment: 16 contiguous halves per lane (p[0..15])
__device__ __forceinline__ v16bf load_frag_k16(const unsigned short* p) {
  union { v16bf v; v8bf h[2]; } f;
  f.h[0] = *(const v8bf*)(p);
  f.h[1] = *(const v8bf*)(p + 8);
  return f.v;
}
// B-style fragment: 16 contiguous halves per lane
__device__ __forceinline__ v16bf load_frag_row16(const unsigned short* p) {
  union { v16bf v; v8bf h[2]; } f;
  f.h[0] = *(const v8bf*)(p);
  f.h[1] = *(const v8bf*)(p + 8);
  return f.v;
}

// Async DMA one 16-byte chunk global -> LDS (ASYNCcnt tracked).
__device__ __forceinline__ void async_b128_to_lds(unsigned short* lds_dst,
                                                  const unsigned short* gsrc) {
  unsigned int loff = (unsigned int)(size_t)lds_dst;  // low 32b of flat = LDS offset
  unsigned long long ga = (unsigned long long)(size_t)gsrc;
  asm volatile("global_load_async_to_lds_b128 %0, %1, off"
               :: "v"(loff), "v"(ga)
               : "memory");
}

// ---------------------------------------------------------------------------
// Embedding: x = tok_emb[idx] + prop @ prop_w^T + prop_b + pos_emb
// ---------------------------------------------------------------------------
__global__ __launch_bounds__(256) void embed_kernel(
    float* __restrict__ x, const int* __restrict__ idx,
    const float* __restrict__ prop, const float* __restrict__ tok_emb,
    const float* __restrict__ prop_w, const float* __restrict__ prop_b,
    const float* __restrict__ pos_emb) {
  int bt = blockIdx.x;
  int b = bt >> 10;
  int t = bt & 1023;
  int tok = idx[bt];
  float p0 = prop[b * 3 + 0], p1 = prop[b * 3 + 1], p2 = prop[b * 3 + 2];
  for (int c = threadIdx.x; c < C_DIM; c += 256) {
    float pw = prop_w[c * 3 + 0] * p0 + prop_w[c * 3 + 1] * p1 + prop_w[c * 3 + 2] * p2;
    x[bt * C_DIM + c] = tok_emb[tok * C_DIM + c] + pw + prop_b[c] + pos_emb[t * C_DIM + c];
  }
}

// ---------------------------------------------------------------------------
// LayerNorm (f32 in) -> bf16 out
// ---------------------------------------------------------------------------
__global__ __launch_bounds__(256) void ln_kernel(
    unsigned short* __restrict__ out, const float* __restrict__ x,
    const float* __restrict__ w, const float* __restrict__ b) {
  int row = blockIdx.x;
  const float* xr = x + row * C_DIM;
  float v[4];
  float s = 0.f, ss = 0.f;
#pragma unroll
  for (int i = 0; i < 4; ++i) {
    v[i] = xr[threadIdx.x + i * 256];
    s += v[i];
    ss += v[i] * v[i];
  }
#pragma unroll
  for (int o = 16; o > 0; o >>= 1) {
    s += __shfl_xor(s, o, 32);
    ss += __shfl_xor(ss, o, 32);
  }
  __shared__ float red[16];
  int wv = threadIdx.x >> 5, ln = threadIdx.x & 31;
  if (ln == 0) { red[wv] = s; red[8 + wv] = ss; }
  __syncthreads();
  if (threadIdx.x < 32) {
    float a = (ln < 8) ? red[ln] : 0.f;
    float c = (ln < 8) ? red[8 + ln] : 0.f;
#pragma unroll
    for (int o = 4; o > 0; o >>= 1) {
      a += __shfl_xor(a, o, 32);
      c += __shfl_xor(c, o, 32);
    }
    if (ln == 0) { red[0] = a; red[1] = c; }
  }
  __syncthreads();
  float mean = red[0] * (1.0f / C_DIM);
  float var = red[1] * (1.0f / C_DIM) - mean * mean;
  float rstd = rsqrtf(var + 1e-5f);
#pragma unroll
  for (int i = 0; i < 4; ++i) {
    int c = threadIdx.x + i * 256;
    out[row * C_DIM + c] = f32_to_bf16((v[i] - mean) * rstd * w[c] + b[c]);
  }
}

// ---------------------------------------------------------------------------
// fp32 -> bf16 converter (vectorized x4)
// ---------------------------------------------------------------------------
struct __align__(8) us4 { unsigned short x, y, z, w; };

__global__ __launch_bounds__(256) void cvt_kernel(
    unsigned short* __restrict__ dst, const float* __restrict__ src, int n4) {
  int i = blockIdx.x * blockDim.x + threadIdx.x;
  int stride = gridDim.x * blockDim.x;
  for (; i < n4; i += stride) {
    float4 f = ((const float4*)src)[i];
    us4 o;
    o.x = f32_to_bf16(f.x);
    o.y = f32_to_bf16(f.y);
    o.z = f32_to_bf16(f.z);
    o.w = f32_to_bf16(f.w);
    ((us4*)dst)[i] = o;
  }
}

// ---------------------------------------------------------------------------
// WMMA GEMM: Y[M,N] = X[M,K] @ W[N,K]^T (+bias) (+GELU) (+residual into Yf)
// flags: bit0 = exact GELU, bit1 = accumulate into Yf, bit2 = store Ybf
//        transposed as [B,H,D,T] (for V)
// Block: 128 threads = 4 waves in 2x2; block tile 128x128; wave tile 64x64
// (4x4 WMMA grid -> 16 WMMAs per 16 ds_load_b128 = 512B LDS traffic/WMMA).
// Tiles staged in LDS via global_load_async_to_lds_b128, double-buffered.
// ---------------------------------------------------------------------------
#define LDS_STRIDE 40  // 32 halves of data + 8 halves pad (80B rows)

// Each thread owns one 32-half tile row: copy it as 4 async b128 chunks.
__device__ __forceinline__ void async_copy_row(unsigned short* lds_row,
                                               const unsigned short* grow) {
#pragma unroll
  for (int i = 0; i < 4; ++i)
    async_b128_to_lds(lds_row + i * 8, grow + i * 8);
}

__global__ __launch_bounds__(128) void wmma_gemm_bf16(
    const unsigned short* __restrict__ X, const unsigned short* __restrict__ W,
    const float* __restrict__ bias, float* __restrict__ Yf,
    unsigned short* __restrict__ Ybf, int N, int K, int flags) {
  const int tid = threadIdx.x;
  const int lane = tid & 31;
  const int wv = tid >> 5;
  const int wm = wv & 1;
  const int wn = wv >> 1;
  const int lh = lane >> 4;
  const int ln = lane & 15;
  const int rowBase = blockIdx.y * 128 + wm * 64;
  const int colBase = blockIdx.x * 128 + wn * 64;

  __shared__ __align__(16) unsigned short ldsX[2][128 * LDS_STRIDE];
  __shared__ __align__(16) unsigned short ldsW[2][128 * LDS_STRIDE];

  const v8f vz = {0.f, 0.f, 0.f, 0.f, 0.f, 0.f, 0.f, 0.f};
  v8f acc[4][4];
#pragma unroll
  for (int s = 0; s < 4; ++s)
#pragma unroll
    for (int t = 0; t < 4; ++t) acc[s][t] = vz;

  const unsigned short* gx = X + (blockIdx.y * 128 + tid) * K;
  const unsigned short* gw = W + (blockIdx.x * 128 + tid) * K;
  unsigned short* lx0 = &ldsX[0][0] + tid * LDS_STRIDE;
  unsigned short* lx1 = &ldsX[1][0] + tid * LDS_STRIDE;
  unsigned short* lw0 = &ldsW[0][0] + tid * LDS_STRIDE;
  unsigned short* lw1 = &ldsW[1][0] + tid * LDS_STRIDE;

  // Prologue: fill buffer 0.
  async_copy_row(lx0, gx);
  async_copy_row(lw0, gw);
  asm volatile("s_wait_asynccnt 0x0" ::: "memory");
  __syncthreads();

  int buf = 0;
  for (int k = 0; k < K; k += 32) {
    if (k + 32 < K) {
      async_copy_row(buf ? lx0 : lx1, gx + k + 32);
      async_copy_row(buf ? lw0 : lw1, gw + k + 32);
    }

    const unsigned short* pA = &ldsX[buf][0] + (wm * 64 + ln) * LDS_STRIDE + lh * 16;
    const unsigned short* pB = &ldsW[buf][0] + (wn * 64 + ln) * LDS_STRIDE + lh * 16;
    v16bf b0 = load_frag_row16(pB);
    v16bf b1 = load_frag_row16(pB + 16 * LDS_STRIDE);
    v16bf b2 = load_frag_row16(pB + 32 * LDS_STRIDE);
    v16bf b3 = load_frag_row16(pB + 48 * LDS_STRIDE);
#pragma unroll
    for (int s = 0; s < 4; ++s) {
      v16bf a = load_frag_k16(pA + s * 16 * LDS_STRIDE);
      acc[s][0] = WMMA_BF16(a, b0, acc[s][0]);
      acc[s][1] = WMMA_BF16(a, b1, acc[s][1]);
      acc[s][2] = WMMA_BF16(a, b2, acc[s][2]);
      acc[s][3] = WMMA_BF16(a, b3, acc[s][3]);
    }

    // Next-buffer fills must land, and all waves must be done reading `buf`.
    asm volatile("s_wait_asynccnt 0x0" ::: "memory");
    __syncthreads();
    buf ^= 1;
  }

#pragma unroll
  for (int s = 0; s < 4; ++s)
#pragma unroll
    for (int t = 0; t < 4; ++t)
#pragma unroll
      for (int r = 0; r < 8; ++r) {
        int m = rowBase + s * 16 + r + 8 * lh;
        int n = colBase + t * 16 + ln;
        float v = acc[s][t][r];
        if (bias) v += bias[n];
        if (flags & 1) v = 0.5f * v * (1.0f + erff(v * 0.70710678118654752f));
        int idx = m * N + n;
        if (flags & 2) v += Yf[idx];
        if (Yf) Yf[idx] = v;
        if (Ybf) {
          int oi = (flags & 4) ? (((m >> 10) * 1024 + n) * 1024 + (m & 1023)) : idx;
          Ybf[oi] = f32_to_bf16(v);
        }
      }
}

// ---------------------------------------------------------------------------
// Causal flash attention, one (batch, head) per blockIdx.x, 128 queries per
// blockIdx.y; each of 8 waves owns 16 query rows. q/k: [B,T,C]; v: [B,H,D,T].
// ---------------------------------------------------------------------------
__global__ __launch_bounds__(256) void flash_attn_wmma(
    unsigned short* __restrict__ y_bf, const unsigned short* __restrict__ q_bf,
    const unsigned short* __restrict__ k_bf,
    const unsigned short* __restrict__ vt_bf) {
  const int b = blockIdx.x >> 4;
  const int hh = blockIdx.x & 15;
  const int wv = threadIdx.x >> 5;
  const int lane = threadIdx.x & 31;
  const int lh = lane >> 4;
  const int ln = lane & 15;
  const int q0 = blockIdx.y * 128 + wv * 16;

  __shared__ __align__(16) unsigned short lds_p[8][16 * 32];

  const unsigned short* qp =
      q_bf + (b * T_DIM + q0 + ln) * C_DIM + hh * D_DIM + lh * 16;
  v16bf aQ0 = load_frag_k16(qp);
  v16bf aQ1 = load_frag_k16(qp + 32);

  const v8f vz = {0.f, 0.f, 0.f, 0.f, 0.f, 0.f, 0.f, 0.f};
  v8f o[4];
#pragma unroll
  for (int t4 = 0; t4 < 4; ++t4) o[t4] = vz;
  float rmax[8], rsum[8];
#pragma unroll
  for (int r = 0; r < 8; ++r) { rmax[r] = -3.0e38f; rsum[r] = 0.f; }

  const float scale = 0.125f;  // 1/sqrt(D), D=64

  for (int j0 = 0; j0 < q0 + 16; j0 += 32) {
    v8f S0 = vz, S1 = vz;
    {
      const unsigned short* kp0 =
          k_bf + (b * T_DIM + j0 + ln) * C_DIM + hh * D_DIM + lh * 16;
      const unsigned short* kp1 = kp0 + 16 * C_DIM;
      S0 = WMMA_BF16(aQ0, load_frag_row16(kp0), S0);
      S0 = WMMA_BF16(aQ1, load_frag_row16(kp0 + 32), S0);
      S1 = WMMA_BF16(aQ0, load_frag_row16(kp1), S1);
      S1 = WMMA_BF16(aQ1, load_frag_row16(kp1 + 32), S1);
    }
    float p0[8], p1[8];
#pragma unroll
    for (int r = 0; r < 8; ++r) {
      int qi = q0 + r + 8 * lh;
      float v0 = S0[r] * scale;
      float v1 = S1[r] * scale;
      if (j0 + ln > qi) v0 = -3.0e38f;
      if (j0 + 16 + ln > qi) v1 = -3.0e38f;
      float mx = fmaxf(v0, v1);
#pragma unroll
      for (int os = 1; os < 16; os <<= 1) mx = fmaxf(mx, __shfl_xor(mx, os, 16));
      mx = fmaxf(mx, rmax[r]);
      float corr = __expf(rmax[r] - mx);
      rmax[r] = mx;
      float e0 = (v0 < -1.0e37f) ? 0.f : __expf(v0 - mx);
      float e1 = (v1 < -1.0e37f) ? 0.f : __expf(v1 - mx);
      float sm = e0 + e1;
#pragma unroll
      for (int os = 1; os < 16; os <<= 1) sm += __shfl_xor(sm, os, 16);
      rsum[r] = rsum[r] * corr + sm;
      p0[r] = e0;
      p1[r] = e1;
#pragma unroll
      for (int t4 = 0; t4 < 4; ++t4) o[t4][r] *= corr;
    }
    // Re-layout P (C-layout f32) -> A-frag bf16 via per-wave LDS scratch.
    unsigned short* pl = &lds_p[wv][0];
#pragma unroll
    for (int r = 0; r < 8; ++r) {
      int m = r + 8 * lh;
      pl[m * 32 + ln] = f32_to_bf16(p0[r]);
      pl[m * 32 + 16 + ln] = f32_to_bf16(p1[r]);
    }
    asm volatile("s_wait_dscnt 0" ::: "memory");
    v16bf aP = load_frag_k16(pl + ln * 32 + lh * 16);

    const unsigned short* vp =
        vt_bf + (b * 1024 + hh * D_DIM + ln) * T_DIM + j0 + lh * 16;
#pragma unroll
    for (int t4 = 0; t4 < 4; ++t4)
      o[t4] = WMMA_BF16(aP, load_frag_row16(vp + t4 * 16 * T_DIM), o[t4]);
  }

#pragma unroll
  for (int r = 0; r < 8; ++r) rsum[r] = 1.0f / rsum[r];
#pragma unroll
  for (int t4 = 0; t4 < 4; ++t4)
#pragma unroll
    for (int r = 0; r < 8; ++r) {
      int m = q0 + r + 8 * lh;
      y_bf[(b * T_DIM + m) * C_DIM + hh * D_DIM + t4 * 16 + ln] =
          f32_to_bf16(o[t4][r] * rsum[r]);
    }
}

// ---------------------------------------------------------------------------
// Host orchestration
// ---------------------------------------------------------------------------
extern "C" void kernel_launch(void* const* d_in, const int* in_sizes, int n_in,
                              void* d_out, int out_size, void* d_ws,
                              size_t ws_size, hipStream_t stream) {
  (void)in_sizes; (void)n_in; (void)out_size; (void)ws_size;
  const int* idx = (const int*)d_in[0];
  const float* prop = (const float*)d_in[1];
  const float* tok_emb = (const float*)d_in[2];
  const float* prop_w = (const float*)d_in[3];
  const float* prop_b = (const float*)d_in[4];
  const float* pos_emb = (const float*)d_in[5];
  const float* ln1_w = (const float*)d_in[6];
  const float* ln1_b = (const float*)d_in[7];
  const float* Wq = (const float*)d_in[8];
  const float* bq = (const float*)d_in[9];
  const float* Wk = (const float*)d_in[10];
  const float* bk = (const float*)d_in[11];
  const float* Wv = (const float*)d_in[12];
  const float* bv = (const float*)d_in[13];
  const float* Wo = (const float*)d_in[14];
  const float* bo = (const float*)d_in[15];
  const float* ln2_w = (const float*)d_in[16];
  const float* ln2_b = (const float*)d_in[17];
  const float* W1 = (const float*)d_in[18];
  const float* b1 = (const float*)d_in[19];
  const float* W2 = (const float*)d_in[20];
  const float* b2 = (const float*)d_in[21];
  const float* lnf_w = (const float*)d_in[22];
  const float* lnf_b = (const float*)d_in[23];
  const float* head_w = (const float*)d_in[24];

  char* ws = (char*)d_ws;
  const size_t MB = 1u << 20;
  float* x = (float*)(ws);                                  // 16 MB f32 residual
  unsigned short* h_bf = (unsigned short*)(ws + 16 * MB);   // 8 MB
  unsigned short* q_bf = (unsigned short*)(ws + 24 * MB);   // 8 MB
  unsigned short* k_bf = (unsigned short*)(ws + 32 * MB);   // 8 MB
  unsigned short* vt_bf = (unsigned short*)(ws + 40 * MB);  // 8 MB [B,H,D,T]
  unsigned short* y_bf = (unsigned short*)(ws + 48 * MB);   // 8 MB
  unsigned short* ff_bf = (unsigned short*)(ws + 56 * MB);  // 32 MB
  unsigned short* wbuf = (unsigned short*)(ws + 88 * MB);   // 8 MB weight staging

  auto cvt = [&](unsigned short* dst, const float* src, int n) {
    int n4 = n >> 2;
    int blocks = (n4 + 255) / 256;
    if (blocks > 2048) blocks = 2048;
    cvt_kernel<<<blocks, 256, 0, stream>>>(dst, src, n4);
  };
  auto gemm = [&](const unsigned short* X, const unsigned short* W,
                  const float* bias, float* Yf, unsigned short* Ybf, int N,
                  int K, int flags) {
    dim3 grid(N / 128, BT / 128);
    wmma_gemm_bf16<<<grid, 128, 0, stream>>>(X, W, bias, Yf, Ybf, N, K, flags);
  };

  embed_kernel<<<BT, 256, 0, stream>>>(x, idx, prop, tok_emb, prop_w, prop_b,
                                       pos_emb);

  for (int i = 0; i < L_DIM; ++i) {
    ln_kernel<<<BT, 256, 0, stream>>>(h_bf, x, ln1_w + i * C_DIM,
                                      ln1_b + i * C_DIM);

    cvt(wbuf, Wq + i * C_DIM * C_DIM, C_DIM * C_DIM);
    gemm(h_bf, wbuf, bq + i * C_DIM, nullptr, q_bf, C_DIM, C_DIM, 0);
    cvt(wbuf, Wk + i * C_DIM * C_DIM, C_DIM * C_DIM);
    gemm(h_bf, wbuf, bk + i * C_DIM, nullptr, k_bf, C_DIM, C_DIM, 0);
    cvt(wbuf, Wv + i * C_DIM * C_DIM, C_DIM * C_DIM);
    gemm(h_bf, wbuf, bv + i * C_DIM, nullptr, vt_bf, C_DIM, C_DIM, 4);

    flash_attn_wmma<<<dim3(B_DIM * H_DIM, T_DIM / 128), 256, 0, stream>>>(
        y_bf, q_bf, k_bf, vt_bf);

    cvt(wbuf, Wo + i * C_DIM * C_DIM, C_DIM * C_DIM);
    gemm(y_bf, wbuf, bo + i * C_DIM, x, nullptr, C_DIM, C_DIM, 2);

    ln_kernel<<<BT, 256, 0, stream>>>(h_bf, x, ln2_w + i * C_DIM,
                                      ln2_b + i * C_DIM);

    cvt(wbuf, W1 + i * FF_DIM * C_DIM, FF_DIM * C_DIM);
    gemm(h_bf, wbuf, b1 + i * FF_DIM, nullptr, ff_bf, FF_DIM, C_DIM, 1);
    cvt(wbuf, W2 + i * C_DIM * FF_DIM, C_DIM * FF_DIM);
    gemm(ff_bf, wbuf, b2 + i * C_DIM, x, nullptr, C_DIM, FF_DIM, 2);
  }

  ln_kernel<<<BT, 256, 0, stream>>>(h_bf, x, lnf_w, lnf_b);
  cvt(wbuf, head_w, V_DIM * C_DIM);
  gemm(h_bf, wbuf, nullptr, (float*)d_out, nullptr, V_DIM, C_DIM, 0);
}
